// EmbedMatcher_26079041422149
// MI455X (gfx1250) — matrix operand order repaired and checked by reading, compiled, not run on hardware
//
#include <hip/hip_runtime.h>
#include <math.h>

typedef __attribute__((ext_vector_type(16))) _Float16 v16h;
typedef __attribute__((ext_vector_type(8)))  float    v8f;

#define DD   128
#define MNB  128
#define KTB  64
#define KSEL 32
#define BQ   2048
#define FEW  5
#define DM   256
#define HID  512
#define G4   2048   // 4*HID

__device__ __forceinline__ float dev_sigmoid(float x) { return 1.f / (1.f + expf(-x)); }

// Full-block (256 thread / 8 wave) sum. Safe to call repeatedly (leading sync).
__device__ __forceinline__ float block_sum(float v, float* red) {
#pragma unroll
  for (int o = 16; o > 0; o >>= 1) v += __shfl_xor(v, o, 32);
  int lane = threadIdx.x & 31, wv = threadIdx.x >> 5;
  __syncthreads();
  if (lane == 0) red[wv] = v;
  __syncthreads();
  float s = 0.f;
#pragma unroll
  for (int i = 0; i < 8; i++) s += red[i];
  return s;
}

// ---------------------------------------------------------------------------
// Neighbor encoder: one block per entity row.
// cos-sim -> stable top-32 -> mean of selected rows -> GEMV(256->128) -> gate
// ---------------------------------------------------------------------------
__global__ __launch_bounds__(256) void neighbor_kernel(
    const int* __restrict__ ids_base, int ids_stride, int ids_off,
    const int* __restrict__ conn,       // [N, 128, 2]
    const int* __restrict__ knn_tab,    // [(NSYM+1), 64]
    const float* __restrict__ emb,      // [(NSYM+1), 128]
    const float* __restrict__ gcn_w,    // [128, 256]
    const float* __restrict__ gcn_wb,
    const float* __restrict__ gcn_bb,
    const float* __restrict__ gate_w,   // [256]
    const float* __restrict__ gate_b,   // [1]
    float* __restrict__ outp, int ldo, int out_off)
{
  __shared__ float cen[DD];
  __shared__ float sim[MNB];
  __shared__ float srel[DD], sent[DD], skent[DD];
  __shared__ float sA[DD], sB[DD];
  __shared__ int   flags[MNB];
  __shared__ float red[8];
  __shared__ float ncen_s, alpha_s;

  const int n = blockIdx.x;
  const int tid = threadIdx.x;
  const int lane = tid & 31, wv = tid >> 5;
  const int id = ids_base[(size_t)n * ids_stride + ids_off];

  float cv = 0.f;
  if (tid < DD) { cv = emb[(size_t)id * DD + tid]; cen[tid] = cv; }
  float ss = block_sum(cv * cv, red);
  if (tid == 0) ncen_s = sqrtf(ss);
  __syncthreads();
  const float nc = ncen_s;

  // ---- struct branch sims: 8 waves sweep m = 0..127 ----
  for (int m = wv; m < MNB; m += 8) {
    int eid = conn[((size_t)n * MNB + m) * 2 + 1];
    const float* e = emb + (size_t)eid * DD;
    float dt = 0.f, nn = 0.f;
#pragma unroll
    for (int i = 0; i < 4; i++) {
      float ev = e[lane + 32 * i];
      dt += cen[lane + 32 * i] * ev;
      nn += ev * ev;
    }
#pragma unroll
    for (int o = 16; o > 0; o >>= 1) { dt += __shfl_xor(dt, o, 32); nn += __shfl_xor(nn, o, 32); }
    if (lane == 0) sim[m] = dt / fmaxf(sqrtf(nn) * nc, 1e-8f);
  }
  __syncthreads();

  // ---- stable top-32 of 128 (rank < K; tie -> lower index), deterministic ----
  if (tid < MNB) {
    float sv = sim[tid];
    int cnt = 0;
    for (int j = 0; j < MNB; j++) {
      float sj = sim[j];
      cnt += (sj > sv) || (sj == sv && j < tid);
    }
    flags[tid] = (cnt < KSEL) ? 1 : 0;
  }
  __syncthreads();

  // ---- accumulate selected rel/ent rows (fixed order -> deterministic) ----
  float racc = 0.f, eacc = 0.f;
  for (int m = 0; m < MNB; m++) {
    if (flags[m]) {
      int rid = conn[((size_t)n * MNB + m) * 2 + 0];
      int eid = conn[((size_t)n * MNB + m) * 2 + 1];
      if (tid < DD) {
        racc += emb[(size_t)rid * DD + tid];
        eacc += emb[(size_t)eid * DD + tid];
      }
    }
  }
  if (tid < DD) { srel[tid] = racc; sent[tid] = eacc; }
  __syncthreads();

  // ---- knn branch sims over 64 table neighbors ----
  for (int m = wv; m < KTB; m += 8) {
    int eid = knn_tab[(size_t)id * KTB + m];
    const float* e = emb + (size_t)eid * DD;
    float dt = 0.f, nn = 0.f;
#pragma unroll
    for (int i = 0; i < 4; i++) {
      float ev = e[lane + 32 * i];
      dt += cen[lane + 32 * i] * ev;
      nn += ev * ev;
    }
#pragma unroll
    for (int o = 16; o > 0; o >>= 1) { dt += __shfl_xor(dt, o, 32); nn += __shfl_xor(nn, o, 32); }
    if (lane == 0) sim[m] = dt / fmaxf(sqrtf(nn) * nc, 1e-8f);
  }
  __syncthreads();
  if (tid < KTB) {
    float sv = sim[tid];
    int cnt = 0;
    for (int j = 0; j < KTB; j++) {
      float sj = sim[j];
      cnt += (sj > sv) || (sj == sv && j < tid);
    }
    flags[tid] = (cnt < KSEL) ? 1 : 0;
  }
  __syncthreads();
  float kacc = 0.f;
  for (int m = 0; m < KTB; m++) {
    if (flags[m]) {
      int eid = knn_tab[(size_t)id * KTB + m];
      if (tid < DD) kacc += emb[(size_t)eid * DD + tid];
    }
  }
  if (tid < DD) skent[tid] = kacc;
  __syncthreads();

  // ---- struct / knn_mean: mean-then-GEMV (linearity of GEMM over k) ----
  if (tid < DD) {
    const float inv = 1.f / (float)KSEL;
    float b = gcn_wb[tid] + gcn_bb[tid];
    const float* wrow = gcn_w + (size_t)tid * 256;
    float a1 = b, a2 = b;
    for (int j = 0; j < DD; j++) a1 += srel[j] * inv * wrow[j];
    for (int j = 0; j < DD; j++) {
      float wj = wrow[DD + j];
      a1 += sent[j]  * inv * wj;
      a2 += skent[j] * inv * wj;   // krel is the zero pad row -> only ent half
    }
    sA[tid] = tanhf(a1);
    sB[tid] = tanhf(a2);
  }
  __syncthreads();

  float part = (tid < DD ? sA[tid] : sB[tid - DD]) * gate_w[tid];
  float g = block_sum(part, red);
  if (tid == 0) alpha_s = dev_sigmoid(g + gate_b[0]);
  __syncthreads();
  if (tid < DD) {
    float a = alpha_s;
    outp[(size_t)n * ldo + out_off + tid] = (1.f - a) * sA[tid] + a * sB[tid];
  }
}

// ---------------------------------------------------------------------------
// Support encoder on the 5 support rows + mean + l2-normalized copy.
// ---------------------------------------------------------------------------
__global__ __launch_bounds__(256) void support_small_kernel(
    const float* __restrict__ sn,
    const float* __restrict__ p1_w, const float* __restrict__ p1_b,
    const float* __restrict__ p2_w, const float* __restrict__ p2_b,
    const float* __restrict__ ln_g, const float* __restrict__ ln_b,
    float* __restrict__ support_g, float* __restrict__ sgn)
{
  __shared__ float x[DM], h1[HID], zz[DM], acc[DM];
  __shared__ float red[8];
  const int tid = threadIdx.x;
  acc[tid] = 0.f;
  for (int row = 0; row < FEW; row++) {
    __syncthreads();
    x[tid] = sn[row * DM + tid];
    __syncthreads();
#pragma unroll
    for (int rep = 0; rep < 2; rep++) {
      int o = tid + 256 * rep;
      float d = p1_b[o];
      const float* w = p1_w + (size_t)o * DM;
      for (int j = 0; j < DM; j++) d += x[j] * w[j];
      h1[o] = fmaxf(d, 0.f);
    }
    __syncthreads();
    {
      float d = p2_b[tid];
      const float* w = p2_w + (size_t)tid * HID;
      for (int j = 0; j < HID; j++) d += h1[j] * w[j];
      zz[tid] = d + x[tid];
    }
    float mu  = block_sum(zz[tid], red) * (1.f / 256.f);
    float df  = zz[tid] - mu;
    float var = block_sum(df * df, red) * (1.f / 255.f);   // unbiased (ddof=1)
    float v = df / (sqrtf(var) + 1e-3f) * ln_g[tid] + ln_b[tid];
    acc[tid] += v * (1.f / (float)FEW);
  }
  __syncthreads();
  float sg = acc[tid];
  support_g[tid] = sg;
  float nrm = sqrtf(block_sum(sg * sg, red));
  sgn[tid] = sg / fmaxf(nrm, 1e-12f);
}

// bias0 = b_ih + b_hh ; bias1 = bias0 + support_g @ w_hh[:,256:512].T
__global__ void bias_kernel(const float* __restrict__ sgv, const float* __restrict__ w_hh,
                            const float* __restrict__ b_ih, const float* __restrict__ b_hh,
                            float* __restrict__ bias0, float* __restrict__ bias1)
{
  int j = blockIdx.x * blockDim.x + threadIdx.x;
  if (j >= G4) return;
  const float* w = w_hh + (size_t)j * HID + DM;
  float s = 0.f;
  for (int k = 0; k < DM; k++) s += sgv[k] * w[k];
  float b0 = b_ih[j] + b_hh[j];
  bias0[j] = b0;
  bias1[j] = b0 + s;
}

__global__ void cvt_f16_kernel(const float* __restrict__ src, _Float16* __restrict__ dst,
                               int rows, int src_ld, int cols)
{
  int i = blockIdx.x * blockDim.x + threadIdx.x;
  if (i >= rows * cols) return;
  int r = i / cols, c = i - r * cols;
  dst[i] = (_Float16)src[(size_t)r * src_ld + c];
}

// ---------------------------------------------------------------------------
// WMMA NT GEMM: C[M,N] = A[M,K] * W[N,K]^T  (v_wmma_f32_16x16x32_f16)
// block = 8 waves; each wave computes NT 16x16 C tiles along N, reusing one
// A fragment per k-step across NT WMMAs (A traffic /NT, wmma density *NT).
// grid (M/16, N/(128*NT)); K is compile-time for pipelining/unroll.
// EPI 0: relu(acc+bias) -> f16     EPI 1: acc+bias+res -> f32
// EPI 2: acc -> f32                EPI 3: acc+addm+bias -> f32
// ---------------------------------------------------------------------------
template<bool AF32, int EPI, int NT, int KK>
__global__ __launch_bounds__(256) void gemm_nt(
    const void* __restrict__ Aptr, int lda,
    const _Float16* __restrict__ W,
    float* __restrict__ Cf, _Float16* __restrict__ Ch, int ldc,
    const float* __restrict__ bias, const float* __restrict__ addm,
    const float* __restrict__ resm, int ldr)
{
  const int lane = threadIdx.x & 31;
  const int wv   = threadIdx.x >> 5;
  const int m0 = blockIdx.x * 16;
  const int n0 = (blockIdx.y * 8 + wv) * (16 * NT);
  const int arow = m0 + (lane & 15);
  const int akb  = (lane >> 4) * 8;    // A lane K base: {0..7,16..23} / {8..15,24..31}
  const int bln  = lane & 15;
  const int bkb  = (lane >> 4) * 8;    // B lane K base: {0..7,16..23} / {8..15,24..31}

  v8f acc[NT] = {};
  for (int k0 = 0; k0 < KK; k0 += 32) {
    v16h a;
    if constexpr (AF32) {
      const float* ap = (const float*)Aptr + (size_t)arow * lda + k0 + akb;
#pragma unroll
      for (int i = 0; i < 8; i++) a[i] = (_Float16)ap[i];
#pragma unroll
      for (int i = 0; i < 8; i++) a[8 + i] = (_Float16)ap[16 + i];
    } else {
      const _Float16* ap = (const _Float16*)Aptr + (size_t)arow * lda + k0 + akb;
#pragma unroll
      for (int i = 0; i < 8; i++) a[i] = ap[i];
#pragma unroll
      for (int i = 0; i < 8; i++) a[8 + i] = ap[16 + i];
    }
#pragma unroll
    for (int t = 0; t < NT; t++) {
      v16h b;
      const _Float16* bp = W + (size_t)(n0 + 16 * t + bln) * KK + k0 + bkb;
#pragma unroll
      for (int i = 0; i < 16; i++) b[i] = bp[(i < 8) ? i : (i + 8)];
      acc[t] = __builtin_amdgcn_wmma_f32_16x16x32_f16(false, a, false, b, (short)0,
                                                      acc[t], false, false);
    }
  }

  const int crow_base = m0 + ((lane >> 4) * 8);
#pragma unroll
  for (int t = 0; t < NT; t++) {
    const int ccol = n0 + 16 * t + bln;
#pragma unroll
    for (int r = 0; r < 8; r++) {
      int crow = crow_base + r;
      float v = acc[t][r];
      if constexpr (EPI == 0) {
        v += bias[ccol]; v = fmaxf(v, 0.f);
        Ch[(size_t)crow * ldc + ccol] = (_Float16)v;
      } else if constexpr (EPI == 1) {
        v += bias[ccol] + resm[(size_t)crow * ldr + ccol];
        Cf[(size_t)crow * ldc + ccol] = v;
      } else if constexpr (EPI == 2) {
        Cf[(size_t)crow * ldc + ccol] = v;
      } else {
        v += addm[(size_t)crow * ldc + ccol] + bias[ccol];
        Cf[(size_t)crow * ldc + ccol] = v;
      }
    }
  }
}

// LayerNorm(z) with unbiased std -> query_g (f32 + f16 copy)
__global__ __launch_bounds__(256) void ln_kernel(const float* __restrict__ Z,
                                                 const float* __restrict__ ln_g,
                                                 const float* __restrict__ ln_b,
                                                 float* __restrict__ qg, _Float16* __restrict__ qg16)
{
  __shared__ float red[8];
  int row = blockIdx.x, tid = threadIdx.x;
  float z = Z[(size_t)row * DM + tid];
  float mu  = block_sum(z, red) * (1.f / 256.f);
  float df  = z - mu;
  float var = block_sum(df * df, red) * (1.f / 255.f);
  float v = df / (sqrtf(var) + 1e-3f) * ln_g[tid] + ln_b[tid];
  qg[(size_t)row * DM + tid] = v;
  qg16[(size_t)row * DM + tid] = (_Float16)v;
}

// LSTM cell elementwise: per (row, j<512)
__global__ void lstm_cell_kernel(const float* __restrict__ G, const float* __restrict__ bias_first,
                                 float* __restrict__ c, const float* __restrict__ qg,
                                 float* __restrict__ h, _Float16* __restrict__ h16, int first)
{
  int idx = blockIdx.x * blockDim.x + threadIdx.x;
  if (idx >= BQ * HID) return;
  int row = idx >> 9, j = idx & 511;
  const float* g = G + (size_t)row * G4;
  float iv = g[j], fv = g[HID + j], gv = g[2 * HID + j], ov = g[3 * HID + j];
  if (first) {
    iv += bias_first[j]; fv += bias_first[HID + j];
    gv += bias_first[2 * HID + j]; ov += bias_first[3 * HID + j];
  }
  float cp = first ? 0.f : c[idx];
  float cn = dev_sigmoid(fv) * cp + dev_sigmoid(iv) * tanhf(gv);
  c[idx] = cn;
  if (j < DM) {
    float hv = qg[(size_t)row * DM + j] + dev_sigmoid(ov) * tanhf(cn);
    h[(size_t)row * DM + j] = hv;
    h16[(size_t)row * DM + j] = (_Float16)hv;
  }
}

// out[row] = dot(h_row, sgn) / max(||h_row||, 1e-12)
__global__ __launch_bounds__(256) void finalize_kernel(const float* __restrict__ h,
                                                       const float* __restrict__ sgn,
                                                       float* __restrict__ out)
{
  __shared__ float red[8];
  int row = blockIdx.x, tid = threadIdx.x;
  float v = h[(size_t)row * DM + tid];
  float ss = block_sum(v * v, red);
  float dp = block_sum(v * sgn[tid], red);
  if (tid == 0) out[row] = dp / fmaxf(sqrtf(ss), 1e-12f);
}

// ---------------------------------------------------------------------------
extern "C" void kernel_launch(void* const* d_in, const int* in_sizes, int n_in,
                              void* d_out, int out_size, void* d_ws, size_t ws_size,
                              hipStream_t stream) {
  (void)in_sizes; (void)n_in; (void)out_size; (void)ws_size;
  const int*   query    = (const int*)d_in[0];
  const int*   support  = (const int*)d_in[1];
  const int*   q_l_conn = (const int*)d_in[2];
  const int*   q_r_conn = (const int*)d_in[4];
  const int*   s_l_conn = (const int*)d_in[6];
  const int*   s_r_conn = (const int*)d_in[8];
  const int*   knn_tab  = (const int*)d_in[10];
  const float* emb      = (const float*)d_in[11];
  const float* gcn_w    = (const float*)d_in[12];
  const float* gcn_wb   = (const float*)d_in[13];
  const float* gcn_bb   = (const float*)d_in[14];
  const float* gate_w   = (const float*)d_in[15];
  const float* gate_b   = (const float*)d_in[16];
  const float* p1_w     = (const float*)d_in[17];
  const float* p1_b     = (const float*)d_in[18];
  const float* p2_w     = (const float*)d_in[19];
  const float* p2_b     = (const float*)d_in[20];
  const float* ln_g     = (const float*)d_in[21];
  const float* ln_b     = (const float*)d_in[22];
  const float* w_ih     = (const float*)d_in[23];
  const float* w_hh     = (const float*)d_in[24];
  const float* b_ih     = (const float*)d_in[25];
  const float* b_hh     = (const float*)d_in[26];
  float* out = (float*)d_out;

  // ---- workspace carve ----
  char* p = (char*)d_ws;
  auto carve = [&](size_t bytes) { char* r = p; p += (bytes + 255) & ~(size_t)255; return r; };
  float*    qn     = (float*)   carve((size_t)BQ * DM * 4);
  float*    sn     = (float*)   carve((size_t)FEW * DM * 4);
  float*    supg   = (float*)   carve(DM * 4);
  float*    sgn    = (float*)   carve(DM * 4);
  float*    bias0  = (float*)   carve(G4 * 4);
  float*    bias1  = (float*)   carve(G4 * 4);
  _Float16* p1w16  = (_Float16*)carve((size_t)HID * DM * 2);
  _Float16* p2w16  = (_Float16*)carve((size_t)DM * HID * 2);
  _Float16* wih16  = (_Float16*)carve((size_t)G4 * DM * 2);
  _Float16* whhL16 = (_Float16*)carve((size_t)G4 * DM * 2);
  _Float16* H116   = (_Float16*)carve((size_t)BQ * HID * 2);
  float*    Z      = (float*)   carve((size_t)BQ * DM * 4);
  float*    qg     = (float*)   carve((size_t)BQ * DM * 4);
  _Float16* qg16   = (_Float16*)carve((size_t)BQ * DM * 2);
  float*    QW     = (float*)   carve((size_t)BQ * G4 * 4);
  float*    Gbuf   = (float*)   carve((size_t)BQ * G4 * 4);
  float*    cbuf   = (float*)   carve((size_t)BQ * HID * 4);
  float*    hbuf   = (float*)   carve((size_t)BQ * DM * 4);
  _Float16* h16    = (_Float16*)carve((size_t)BQ * DM * 2);

  // ---- weight f16 conversion ----
  auto cvt = [&](const float* src, _Float16* dst, int rows, int ld, int cols) {
    int n = rows * cols;
    cvt_f16_kernel<<<(n + 255) / 256, 256, 0, stream>>>(src, dst, rows, ld, cols);
  };
  cvt(p1_w, p1w16, HID, DM, DM);
  cvt(p2_w, p2w16, DM, HID, HID);
  cvt(w_ih, wih16, G4, DM, DM);
  cvt(w_hh, whhL16, G4, HID, DM);   // left half of w_hh

  // ---- neighbor encoders -> qn [2048,256], sn [5,256] ----
  neighbor_kernel<<<BQ, 256, 0, stream>>>(query, 2, 0, q_l_conn, knn_tab, emb,
      gcn_w, gcn_wb, gcn_bb, gate_w, gate_b, qn, DM, 0);
  neighbor_kernel<<<BQ, 256, 0, stream>>>(query, 2, 1, q_r_conn, knn_tab, emb,
      gcn_w, gcn_wb, gcn_bb, gate_w, gate_b, qn, DM, DD);
  neighbor_kernel<<<FEW, 256, 0, stream>>>(support, 2, 0, s_l_conn, knn_tab, emb,
      gcn_w, gcn_wb, gcn_bb, gate_w, gate_b, sn, DM, 0);
  neighbor_kernel<<<FEW, 256, 0, stream>>>(support, 2, 1, s_r_conn, knn_tab, emb,
      gcn_w, gcn_wb, gcn_bb, gate_w, gate_b, sn, DM, DD);

  // ---- support side: support_g, normalized copy, LSTM bias folding ----
  support_small_kernel<<<1, 256, 0, stream>>>(sn, p1_w, p1_b, p2_w, p2_b, ln_g, ln_b, supg, sgn);
  bias_kernel<<<G4 / 256, 256, 0, stream>>>(supg, w_hh, b_ih, b_hh, bias0, bias1);

  // ---- query support encoder (WMMA) ----
  // H1 = relu(qn @ p1_w.T + p1_b)  -> f16   (M=2048,N=512,K=256, NT=4)
  gemm_nt<true, 0, 4, DM><<<dim3(BQ / 16, HID / 512), 256, 0, stream>>>(
      qn, DM, p1w16, nullptr, H116, HID, p1_b, nullptr, nullptr, 0);
  // Z = H1 @ p2_w.T + p2_b + qn            (M=2048,N=256,K=512, NT=2)
  gemm_nt<false, 1, 2, HID><<<dim3(BQ / 16, DM / 256), 256, 0, stream>>>(
      H116, HID, p2w16, Z, nullptr, DM, p2_b, nullptr, qn, DM);
  // query_g = LN(Z)
  ln_kernel<<<BQ, 256, 0, stream>>>(Z, ln_g, ln_b, qg, qg16);

  // ---- attLSTM (softmax over size-1 axis == 1 -> r == support_g const) ----
  // QW = query_g @ w_ih.T  (step-invariant)  (M=2048,N=2048,K=256, NT=4)
  gemm_nt<false, 2, 4, DM><<<dim3(BQ / 16, G4 / 512), 256, 0, stream>>>(
      qg16, DM, wih16, QW, nullptr, G4, nullptr, nullptr, nullptr, 0);
  // step 0: h_r = 0  ->  G = QW + bias0, c = 0
  lstm_cell_kernel<<<(BQ * HID) / 256, 256, 0, stream>>>(QW, bias0, cbuf, qg, hbuf, h16, 1);
  // steps 1..3: G = h @ w_hh[:, :256].T + QW + bias1
  for (int t = 1; t < 4; t++) {
    gemm_nt<false, 3, 4, DM><<<dim3(BQ / 16, G4 / 512), 256, 0, stream>>>(
        h16, DM, whhL16, Gbuf, nullptr, G4, bias1, QW, nullptr, 0);
    lstm_cell_kernel<<<(BQ * HID) / 256, 256, 0, stream>>>(Gbuf, nullptr, cbuf, qg, hbuf, h16, 0);
  }

  // ---- cosine score ----
  finalize_kernel<<<BQ, 256, 0, stream>>>(hbuf, sgn, out);
}
